// AudioEncoder_67276367725229
// MI455X (gfx1250) — matrix operand order repaired and checked by reading, not compiled, not run on hardware
//
#include <hip/hip_runtime.h>
#include <hip/hip_bf16.h>
#include <math.h>

// ---------------------------------------------------------------------------
// Types for CDNA5 WMMA (wave32, 16x16x32 bf16 -> f32)
// ---------------------------------------------------------------------------
typedef __bf16 bf16t;
typedef __attribute__((ext_vector_type(16))) __bf16 v16bf;
typedef __attribute__((ext_vector_type(8)))  __bf16 v8bf;
typedef __attribute__((ext_vector_type(2)))  __bf16 v2bf;
typedef __attribute__((ext_vector_type(8)))  float  v8f;
typedef __attribute__((ext_vector_type(2)))  float  v2f;

#define TMB 128  // block rows   (4 waves x 32 rows)
#define TNB 64   // block cols   (4 WMMA col tiles)
#define KS  32   // K step (bf16 WMMA depth)
#define KP  40   // padded LDS K-stride (40 halves = 80B, multiple of 16B)

#define CB  4        // batch
#define CT  3000     // mel frames
#define CSP 1500     // frames after stride-2 conv
#define CD  1024     // model dim
#define CH  16       // heads
#define CHD 64       // head dim
#define CFF 4096
#define CNQ 8
#define CBINS 1024
#define NTOK (CB*CSP)          // 6000
#define XSZ  ((long)NTOK*CD)   // 6,144,000

enum { F_BIAS=1, F_GELU=2, F_RESID=4, F_MASK=8, F_CONV=16, F_WT=32 };

__device__ __forceinline__ float gelu_tanh(float x) {
  const float c0 = 0.7978845608028654f;
  return 0.5f * x * (1.f + tanhf(c0 * (x + 0.044715f * x * x * x)));
}

__device__ __forceinline__ v16bf ldfrag(const bf16t* pa, const bf16t* pb) {
  v8bf lo = *(const v8bf*)pa;
  v8bf hi = *(const v8bf*)pb;
  v16bf r;
#pragma unroll
  for (int i = 0; i < 8; ++i) { r[i] = lo[i]; r[8 + i] = hi[i]; }
  return r;
}

// ---------------------------------------------------------------------------
// bf16 WMMA GEMM: C[M,N] = epilogue(A[M,K] @ W[K,N]); K%32==0, N%64==0.
//   F_CONV : A gathered im2col from src (B,T_in,2^lg2C), kernel=3, pad=1
//   F_WT   : W stored transposed (W[n*ldw+k]), e.g. codebooks
// FLAGS is a template parameter so each variant keeps only its own paths
// (keeps VGPR count in the low bank; avoids s_set_vgpr_msb churn).
// 128 threads (4 waves), 128x64 tile; per wave per K-step 8 WMMAs.
// Staging: float2 global loads -> regs, packed v2bf (b32) LDS stores,
// double-buffered with one barrier per K-step.
// ---------------------------------------------------------------------------
template<int FLAGS>
__global__ __launch_bounds__(128, 4) void wmma_gemm(
    const float* __restrict__ A, const float* __restrict__ W,
    const float* __restrict__ bias, const float* __restrict__ resid,
    const float* __restrict__ rowmask, float* __restrict__ C,
    int M, int N, int K, int lda, int ldw,
    const float* __restrict__ src, int T_in, int T_out, int cstride, int lg2C)
{
  __shared__ __align__(16) bf16t As[2][TMB][KP];   // 20,480 B
  __shared__ __align__(16) bf16t Bs[2][TNB][KP];   // 10,240 B

  const int m0 = blockIdx.y * TMB, n0 = blockIdx.x * TNB;
  const int tid = threadIdx.x;
  const int wave = tid >> 5, lane = tid & 31;
  const int half = lane >> 4, l15 = lane & 15;
  const int sub = tid & 15, grp = tid >> 4;        // staging decomposition

  v2f areg[16];   // A: row grp+8j, k = 2*sub
  v2f bregt[8];   // B (WT): row n = grp+8j, k = 2*sub
  float bregn[16];// B (normal): col tid&63, kpair = (tid>>6)+2j

  auto gload = [&](int k0) {
#pragma unroll
    for (int j = 0; j < 16; ++j) {
      int m = m0 + grp + 8 * j;
      int k = k0 + 2 * sub;
      v2f v = {0.f, 0.f};
      if (m < M) {
        if (FLAGS & F_CONV) {
          int b = m / T_out, t = m - b * T_out;
          int ck = k >> lg2C, c = k & ((1 << lg2C) - 1);
          int ts = t * cstride + ck - 1;
          if (ts >= 0 && ts < T_in)
            v = *(const v2f*)&src[((long)(b * T_in + ts) << lg2C) + c];
        } else {
          v = *(const v2f*)&A[(long)m * lda + k];
        }
      }
      areg[j] = v;
    }
    if (FLAGS & F_WT) {
#pragma unroll
      for (int j = 0; j < 8; ++j)
        bregt[j] = *(const v2f*)&W[(long)(n0 + grp + 8 * j) * ldw + k0 + 2 * sub];
    } else {
#pragma unroll
      for (int j = 0; j < 8; ++j) {
        int kp = (tid >> 6) + 2 * j;
        bregn[2 * j]     = W[(long)(k0 + 2 * kp)     * ldw + n0 + (tid & 63)];
        bregn[2 * j + 1] = W[(long)(k0 + 2 * kp + 1) * ldw + n0 + (tid & 63)];
      }
      if (k0 + KS < K)   // gfx1250 global_prefetch_b8
        __builtin_prefetch(&W[(long)(k0 + KS + (tid >> 6)) * ldw + n0 + (tid & 63)], 0, 3);
    }
  };

  auto lstore = [&](int buf) {
#pragma unroll
    for (int j = 0; j < 16; ++j) {
      v2bf p; p[0] = (bf16t)areg[j][0]; p[1] = (bf16t)areg[j][1];
      *(v2bf*)&As[buf][grp + 8 * j][2 * sub] = p;
    }
    if (FLAGS & F_WT) {
#pragma unroll
      for (int j = 0; j < 8; ++j) {
        v2bf p; p[0] = (bf16t)bregt[j][0]; p[1] = (bf16t)bregt[j][1];
        *(v2bf*)&Bs[buf][grp + 8 * j][2 * sub] = p;
      }
    } else {
#pragma unroll
      for (int j = 0; j < 8; ++j) {
        int kp = (tid >> 6) + 2 * j;
        v2bf p; p[0] = (bf16t)bregn[2 * j]; p[1] = (bf16t)bregn[2 * j + 1];
        *(v2bf*)&Bs[buf][tid & 63][2 * kp] = p;
      }
    }
  };

  v8f acc0[4] = {}, acc1[4] = {};
  const int arow0 = wave * 32 + l15, arow1 = arow0 + 16;
  const int ak = half ? 8 : 0;    // A 16x32: K runs {0-7,16-23} / {8-15,24-31}
  const int bk = half ? 8 : 0;    // B 32x16: K runs {0-7,16-23} / {8-15,24-31}

  gload(0);
  lstore(0);
  __syncthreads();

  const int nsteps = K / KS;
  for (int s = 0; s < nsteps; ++s) {
    if (s + 1 < nsteps) gload((s + 1) * KS);

    const int cbuf = s & 1;
    v16bf a0 = ldfrag(&As[cbuf][arow0][ak], &As[cbuf][arow0][ak + 16]);
    v16bf a1 = ldfrag(&As[cbuf][arow1][ak], &As[cbuf][arow1][ak + 16]);
#pragma unroll
    for (int nt = 0; nt < 4; ++nt) {
      const bf16t* bp = &Bs[cbuf][nt * 16 + l15][bk];
      v16bf bfr = ldfrag(bp, bp + 16);
      acc0[nt] = __builtin_amdgcn_wmma_f32_16x16x32_bf16(
          false, a0, false, bfr, (short)0, acc0[nt], false, false);
      acc1[nt] = __builtin_amdgcn_wmma_f32_16x16x32_bf16(
          false, a1, false, bfr, (short)0, acc1[nt], false, false);
    }

    if (s + 1 < nsteps) lstore((s + 1) & 1);
    __syncthreads();
  }

  // ---- epilogue (C layout: VGPR r -> row r + 8*half, col = l15) ----
#pragma unroll
  for (int g = 0; g < 2; ++g) {
    const v8f* ac = g ? acc1 : acc0;
    const int rowbase = m0 + wave * 32 + g * 16 + half * 8;
#pragma unroll
    for (int nt = 0; nt < 4; ++nt) {
      int col = n0 + nt * 16 + l15;                    // N%64==0: in range
      float bv = (FLAGS & F_BIAS) ? bias[col] : 0.f;
#pragma unroll
      for (int r = 0; r < 8; ++r) {
        int row = rowbase + r;
        if (row >= M) continue;
        float y = ac[nt][r] + bv;
        if (FLAGS & F_GELU)  y = gelu_tanh(y);
        if (FLAGS & F_MASK)  y *= rowmask[row];
        if (FLAGS & F_RESID) y += resid[(long)row * N + col];
        C[(long)row * N + col] = y;
      }
    }
  }
}

// ---------------------------------------------------------------------------
// LayerNorm over last dim (D=1024), one block per row
// ---------------------------------------------------------------------------
__global__ __launch_bounds__(256) void layernorm_kernel(
    const float* __restrict__ x, const float* __restrict__ s,
    const float* __restrict__ b, float* __restrict__ y, int Dd)
{
  const long row = blockIdx.x;
  const float* xr = x + row * Dd;
  float ls = 0.f, lq = 0.f;
  for (int i = threadIdx.x; i < Dd; i += 256) { float v = xr[i]; ls += v; lq += v * v; }
  __shared__ float ssum[256], ssq[256];
  ssum[threadIdx.x] = ls; ssq[threadIdx.x] = lq;
  __syncthreads();
  for (int off = 128; off > 0; off >>= 1) {
    if (threadIdx.x < off) {
      ssum[threadIdx.x] += ssum[threadIdx.x + off];
      ssq[threadIdx.x]  += ssq[threadIdx.x + off];
    }
    __syncthreads();
  }
  float mu  = ssum[0] / Dd;
  float var = ssq[0] / Dd - mu * mu;
  float inv = rsqrtf(var + 1e-6f);
  for (int i = threadIdx.x; i < Dd; i += 256)
    y[row * Dd + i] = (xr[i] - mu) * inv * s[i] + b[i];
}

// ---------------------------------------------------------------------------
// RoPE in place on (B,SP,H*HD): thread per (bt,h,j) pair, j in [0,32)
// ---------------------------------------------------------------------------
__global__ __launch_bounds__(256) void rope_kernel(float* __restrict__ x, int total)
{
  int i = blockIdx.x * 256 + threadIdx.x;
  if (i >= total) return;
  int j  = i & 31;
  int h  = (i >> 5) & (CH - 1);
  int bt = i >> 9;
  int t  = bt % CSP;
  float f = (float)t * __expf(-0.28782313662425575f * (float)j);  // 10000^{-j/32}
  float sn = __sinf(f), cs = __cosf(f);
  long base = (long)bt * CD + h * CHD + j;
  float x1 = x[base], x2 = x[base + 32];
  x[base]      = x1 * cs - x2 * sn;
  x[base + 32] = x2 * cs + x1 * sn;
}

// ---------------------------------------------------------------------------
// Windowed attention (+-128), wave-per-query, shuffle softmax
// ---------------------------------------------------------------------------
__global__ __launch_bounds__(256) void attn_kernel(
    const float* __restrict__ Q, const float* __restrict__ Km,
    const float* __restrict__ V, const int* __restrict__ lengths,
    float* __restrict__ CTX)
{
  __shared__ float sS[8][264];
  const int wave = threadIdx.x >> 5, lane = threadIdx.x & 31;
  const int wid = blockIdx.x * 8 + wave;           // [0, B*H*SP)
  const int t = wid % CSP;
  const int h = (wid / CSP) % CH;
  const int b = wid / (CSP * CH);
  const int L = (lengths[b] - 1) / 2 + 1;
  const bool valid = (t < L);
  int klo = valid ? max(0, t - 128) : 0;
  int khi = valid ? min(min(CSP, L), t + 129) : 1;

  const float* qr = Q + ((long)(b * CSP + t) * CD + h * CHD);
  float qv[CHD];
#pragma unroll
  for (int u = 0; u < CHD; ++u) qv[u] = qr[u];

  float lmax = -1e30f;
  for (int kk = klo + lane; kk < khi; kk += 32) {
    const float* kr = Km + ((long)(b * CSP + kk) * CD + h * CHD);
    float d = 0.f;
#pragma unroll 8
    for (int u = 0; u < CHD; ++u) d += qv[u] * kr[u];
    d *= 0.125f;                                    // 1/sqrt(64)
    sS[wave][kk - klo] = d;
    lmax = fmaxf(lmax, d);
  }
  for (int off = 16; off > 0; off >>= 1) lmax = fmaxf(lmax, __shfl_xor(lmax, off, 32));
  float lsum = 0.f;
  for (int kk = klo + lane; kk < khi; kk += 32) {
    float e = __expf(sS[wave][kk - klo] - lmax);
    sS[wave][kk - klo] = e;
    lsum += e;
  }
  for (int off = 16; off > 0; off >>= 1) lsum += __shfl_xor(lsum, off, 32);
  float rinv = 1.f / lsum;
  __syncthreads();

  float a0 = 0.f, a1 = 0.f;
  const int d0 = lane, d1 = lane + 32;
  for (int kk = klo; kk < khi; ++kk) {
    float w = sS[wave][kk - klo];
    const float* vr = V + ((long)(b * CSP + kk) * CD + h * CHD);
    a0 += w * vr[d0];
    a1 += w * vr[d1];
  }
  long ob = (long)(b * CSP + t) * CD + h * CHD;
  CTX[ob + d0] = valid ? a0 * rinv : 0.f;
  CTX[ob + d1] = valid ? a1 * rinv : 0.f;
}

// ---------------------------------------------------------------------------
// RVQ helpers
// ---------------------------------------------------------------------------
__global__ __launch_bounds__(256) void mask_kernel(
    const int* __restrict__ lengths, float* __restrict__ mask,
    float* __restrict__ lens_out)
{
  int i = blockIdx.x * 256 + threadIdx.x;
  if (i < NTOK) {
    int b = i / CSP, t = i - b * CSP;
    int L = (lengths[b] - 1) / 2 + 1;
    mask[i] = (t < L) ? 1.f : 0.f;
  }
  if (i < CB) lens_out[i] = (float)((lengths[i] - 1) / 2 + 1);
}

__global__ __launch_bounds__(256) void cbnorm_kernel(
    const float* __restrict__ cb, float* __restrict__ norms)
{
  const long row = blockIdx.x;    // NQ*BINS rows
  float s = 0.f;
  for (int i = threadIdx.x; i < CD; i += 256) { float v = cb[row * CD + i]; s += v * v; }
  __shared__ float red[256];
  red[threadIdx.x] = s; __syncthreads();
  for (int off = 128; off > 0; off >>= 1) {
    if (threadIdx.x < off) red[threadIdx.x] += red[threadIdx.x + off];
    __syncthreads();
  }
  if (threadIdx.x == 0) norms[row] = red[0];
}

__global__ __launch_bounds__(256) void rvq_init_kernel(
    const float* __restrict__ x, float* __restrict__ resid, float* __restrict__ quant)
{
  long i = (long)blockIdx.x * 256 + threadIdx.x;
  resid[i] = x[i];
  quant[i] = 0.f;
}

__global__ __launch_bounds__(256) void argmin_kernel(
    const float* __restrict__ scores, const float* __restrict__ norms,
    int* __restrict__ idxout, float* __restrict__ codes_out)
{
  const long row = blockIdx.x;    // NTOK rows
  float best = 1e30f; int bidx = 0;
  for (int n = threadIdx.x; n < CBINS; n += 256) {
    float v = norms[n] - 2.f * scores[row * CBINS + n];
    if (v < best) { best = v; bidx = n; }
  }
  __shared__ float bv[256]; __shared__ int bi[256];
  bv[threadIdx.x] = best; bi[threadIdx.x] = bidx;
  __syncthreads();
  for (int off = 128; off > 0; off >>= 1) {
    if (threadIdx.x < off) {
      float ov = bv[threadIdx.x + off]; int oi = bi[threadIdx.x + off];
      if (ov < bv[threadIdx.x] || (ov == bv[threadIdx.x] && oi < bi[threadIdx.x])) {
        bv[threadIdx.x] = ov; bi[threadIdx.x] = oi;
      }
    }
    __syncthreads();
  }
  if (threadIdx.x == 0) { idxout[row] = bi[0]; codes_out[row] = (float)bi[0]; }
}

__global__ __launch_bounds__(256) void rvq_update_kernel(
    const float* __restrict__ cb, const int* __restrict__ idx,
    const float* __restrict__ mask, float* __restrict__ quant,
    float* __restrict__ resid)
{
  long i = (long)blockIdx.x * 256 + threadIdx.x;   // XSZ elements
  int m = (int)(i >> 10), d = (int)(i & (CD - 1));
  float c = cb[((long)idx[m] << 10) + d] * mask[m];
  quant[i] += c;
  resid[i] -= c;
}

// ---------------------------------------------------------------------------
// Host-side orchestration
// ---------------------------------------------------------------------------
template<int FLAGS>
static void launch_gemm(hipStream_t stream,
                        const float* A, const float* W, const float* bias,
                        const float* resid, const float* rm, float* C,
                        int M, int N, int K, int lda, int ldw,
                        const float* src = nullptr, int Tin = 0, int Tout = 0,
                        int cs = 0, int lg2C = 0)
{
  dim3 grid(N / TNB, (M + TMB - 1) / TMB);
  wmma_gemm<FLAGS><<<grid, 128, 0, stream>>>(A, W, bias, resid, rm, C,
                                             M, N, K, lda, ldw,
                                             src, Tin, Tout, cs, lg2C);
}

extern "C" void kernel_launch(void* const* d_in, const int* in_sizes, int n_in,
                              void* d_out, int out_size, void* d_ws, size_t ws_size,
                              hipStream_t stream)
{
  const float* mel     = (const float*)d_in[0];
  const int*   lengths = (const int*)  d_in[1];
  const float* c1w = (const float*)d_in[2]; const float* c1b = (const float*)d_in[3];
  const float* c2w = (const float*)d_in[4]; const float* c2b = (const float*)d_in[5];
  const float* cbs = (const float*)d_in[36];

  float* ws   = (float*)d_ws;
  float* X    = ws;
  float* Hb   = X   + XSZ;
  float* Qb   = Hb  + XSZ;
  float* Kb   = Qb  + XSZ;
  float* Vb   = Kb  + XSZ;
  float* CTX  = Vb  + XSZ;
  float* FFH  = CTX + XSZ;                  // 6000*4096 (also conv1 out, RVQ scores)
  float* maskb = FFH + (long)NTOK * CFF;
  float* norms = maskb + 8192;
  int*   idxb  = (int*)(norms + CNQ * CBINS);

  float* out_x     = (float*)d_out;
  float* out_q     = out_x + XSZ;
  float* out_len   = out_q + XSZ;
  float* out_codes = out_len + CB;

  // conv1: (B*T, 384) @ (384,1024) + gelu -> x1 (in FFH)
  launch_gemm<F_CONV | F_BIAS | F_GELU>(stream, nullptr, c1w, c1b, nullptr, nullptr,
      FFH, CB * CT, CD, 3 * 128, 0, CD, mel, CT, CT, 1, 7);
  // conv2 (stride 2): (B*SP, 3072) @ (3072,1024) + gelu -> X
  launch_gemm<F_CONV | F_BIAS | F_GELU>(stream, nullptr, c2w, c2b, nullptr, nullptr,
      X, NTOK, CD, 3 * CD, 0, CD, FFH, CT, CSP, 2, 10);

  mask_kernel<<<(NTOK + 255) / 256, 256, 0, stream>>>(lengths, maskb, out_len);

  for (int l = 0; l < 2; ++l) {
    const int p0 = 6 + l * 15;
    const float* ln1s = (const float*)d_in[p0 + 0];
    const float* ln1b = (const float*)d_in[p0 + 1];
    const float* qw   = (const float*)d_in[p0 + 2];
    const float* qbv  = (const float*)d_in[p0 + 3];
    const float* kw   = (const float*)d_in[p0 + 4];
    const float* vw   = (const float*)d_in[p0 + 5];
    const float* vbv  = (const float*)d_in[p0 + 6];
    const float* ow   = (const float*)d_in[p0 + 7];
    const float* obv  = (const float*)d_in[p0 + 8];
    const float* ln2s = (const float*)d_in[p0 + 9];
    const float* ln2b = (const float*)d_in[p0 + 10];
    const float* f1w  = (const float*)d_in[p0 + 11];
    const float* f1b  = (const float*)d_in[p0 + 12];
    const float* f2w  = (const float*)d_in[p0 + 13];
    const float* f2b  = (const float*)d_in[p0 + 14];

    layernorm_kernel<<<NTOK, 256, 0, stream>>>(X, ln1s, ln1b, Hb, CD);

    launch_gemm<F_BIAS>(stream, Hb, qw, qbv, nullptr, nullptr, Qb, NTOK, CD, CD, CD, CD);
    launch_gemm<0>     (stream, Hb, kw, nullptr, nullptr, nullptr, Kb, NTOK, CD, CD, CD, CD);
    launch_gemm<F_BIAS>(stream, Hb, vw, vbv, nullptr, nullptr, Vb, NTOK, CD, CD, CD, CD);

    const int rope_total = NTOK * CH * 32;
    rope_kernel<<<(rope_total + 255) / 256, 256, 0, stream>>>(Qb, rope_total);
    rope_kernel<<<(rope_total + 255) / 256, 256, 0, stream>>>(Kb, rope_total);

    attn_kernel<<<(CB * CH * CSP) / 8, 256, 0, stream>>>(Qb, Kb, Vb, lengths, CTX);

    // x = x + mask * (ctx @ o_w + o_b)
    launch_gemm<F_BIAS | F_MASK | F_RESID>(stream, CTX, ow, obv, X, maskb, X,
                                           NTOK, CD, CD, CD, CD);

    layernorm_kernel<<<NTOK, 256, 0, stream>>>(X, ln2s, ln2b, Hb, CD);
    launch_gemm<F_BIAS | F_GELU>(stream, Hb, f1w, f1b, nullptr, nullptr, FFH,
                                 NTOK, CFF, CD, CD, CFF);
    float* dst = (l == 1) ? out_x : X;
    launch_gemm<F_BIAS | F_RESID>(stream, FFH, f2w, f2b, X, nullptr, dst,
                                  NTOK, CD, CFF, CFF, CD);
  }

  // ---- RVQ ----
  cbnorm_kernel<<<CNQ * CBINS, 256, 0, stream>>>(cbs, norms);
  rvq_init_kernel<<<(int)(XSZ / 256), 256, 0, stream>>>(out_x, Hb, out_q);
  for (int q = 0; q < CNQ; ++q) {
    const float* cb = cbs + (long)q * CBINS * CD;
    // scores = residual @ cb^T  (W transposed layout)
    launch_gemm<F_WT>(stream, Hb, cb, nullptr, nullptr, nullptr, FFH,
                      NTOK, CBINS, CD, CD, CD);
    argmin_kernel<<<NTOK, 256, 0, stream>>>(FFH, norms + q * CBINS, idxb,
                                            out_codes + (long)q * NTOK);
    rvq_update_kernel<<<(int)(XSZ / 256), 256, 0, stream>>>(cb, idxb, maskb, out_q, Hb);
  }
}
